// LSTM_unroll_2310692405865
// MI455X (gfx1250) — matrix operand order repaired and checked by reading, not compiled, not run on hardware
//
#include <hip/hip_runtime.h>
#include <hip/hip_bf16.h>

// ---------------------------------------------------------------------------
// 2-layer LSTM + action feedback, MI455X (gfx1250, wave32, WMMA).
// Strategy: weights converted once to bf16 (L2-resident, ~30MB of 192MB L2),
// per-timestep skinny GEMMs run on V_WMMA_F32_16X16X32_BF16 with f32 accum.
// Recurrence carried by stream ordering (256 steps x 6 kernels).
// ---------------------------------------------------------------------------

typedef __attribute__((ext_vector_type(16))) __bf16        v16bf;
typedef __attribute__((ext_vector_type(8)))  float         v8f;
typedef __attribute__((ext_vector_type(4)))  unsigned int  v4u;

#define Bq   64
#define Sq   256
#define Fq   256
#define Hq   1024
#define G4q  4096
#define K0q  1536   // x(256) | action(256) | h0(1024)
#define K1q  2048   // h0(1024) | h1(1024)

__device__ __forceinline__ unsigned short f32_to_bf16_bits(float f) {
  unsigned int u = __float_as_uint(f);
  u += 0x7FFFu + ((u >> 16) & 1u);          // round-to-nearest-even
  return (unsigned short)(u >> 16);
}

// Load a 16-element bf16 fragment from two 16B-aligned halves.
__device__ __forceinline__ v16bf load_frag2(const __bf16* p0, const __bf16* p1) {
  union { v4u u[2]; v16bf v; } f;
  f.u[0] = *(const v4u*)p0;
  f.u[1] = *(const v4u*)p1;
  return f.v;
}

// ---------------------------------------------------------------------------
// GEMM: out[64 x N] = A[64 x K](bf16) * W[N x K](bf16, row-major)^T + bias[N]
// Block = 256 threads = 8 waves. Wave -> (N strip of 16) x (M half of 32).
// Grid.x = N/64. K must be a multiple of 32. f32 accumulate.
// ---------------------------------------------------------------------------
__global__ __launch_bounds__(256)
void gemm64_wmma_bf16(const __bf16* __restrict__ A, int lda,
                      const __bf16* __restrict__ W, int K,
                      const float* __restrict__ bias,
                      float* __restrict__ out, int ldo) {
  const int wave   = threadIdx.x >> 5;
  const int lane   = threadIdx.x & 31;
  const int nstrip = wave & 3;          // 4 N strips per block
  const int mhalf  = wave >> 2;         // M rows 0-31 or 32-63
  const int n0     = blockIdx.x * 64 + nstrip * 16;
  const int col    = lane & 15;
  const int hi     = lane >> 4;

  // B fragment: lane holds column n0+col, K = k0 + 8*hi..+7 and k0 + 16 + 8*hi..+7.
  const __bf16* wp = W + (size_t)(n0 + col) * K + hi * 8;
  // A fragment: lane holds row, K = k0 + 8*hi..+7 and k0 + 16 + 8*hi..+7.
  const __bf16* a0 = A + (size_t)(mhalf * 32 + col) * lda + hi * 8;
  const __bf16* a1 = a0 + (size_t)16 * lda;

  v8f acc0 = {}; v8f acc1 = {};
  for (int k0 = 0; k0 < K; k0 += 32) {
    v16bf bF  = load_frag2(wp + k0, wp + k0 + 16);
    v16bf aF0 = load_frag2(a0 + k0, a0 + k0 + 16);
    v16bf aF1 = load_frag2(a1 + k0, a1 + k0 + 16);
    acc0 = __builtin_amdgcn_wmma_f32_16x16x32_bf16(false, aF0, false, bF,
                                                   (short)0, acc0, false, false);
    acc1 = __builtin_amdgcn_wmma_f32_16x16x32_bf16(false, aF1, false, bF,
                                                   (short)0, acc1, false, false);
  }

  // C/D layout: VGPR r -> M = 8*hi + r, N = col.
  const float bv = bias[n0 + col];
  float* o0 = out + (size_t)(mhalf * 32 + hi * 8) * ldo + (n0 + col);
  float* o1 = o0 + (size_t)16 * ldo;
#pragma unroll
  for (int r = 0; r < 8; ++r) {
    o0[(size_t)r * ldo] = acc0[r] + bv;
    o1[(size_t)r * ldo] = acc1[r] + bv;
  }
}

// ---------------------------------------------------------------------------
// LSTM cell elementwise: gates[64 x 4096] (i|f|g|o), c state update, h -> two
// bf16 staging destinations (next-step GEMM inputs). 65536 threads.
// ---------------------------------------------------------------------------
__global__ void lstm_cell_k(const float* __restrict__ gates,
                            float* __restrict__ c,
                            unsigned short* __restrict__ h1, int ld1,
                            unsigned short* __restrict__ h2, int ld2) {
  int idx = blockIdx.x * blockDim.x + threadIdx.x;   // [0, 64*1024)
  int b = idx >> 10, j = idx & 1023;
  const float* g = gates + (size_t)b * G4q;
  float iv = g[j], fv = g[Hq + j], gv = g[2 * Hq + j], ov = g[3 * Hq + j];
  iv = 1.f / (1.f + __expf(-iv));
  fv = 1.f / (1.f + __expf(-fv));
  ov = 1.f / (1.f + __expf(-ov));
  gv = tanhf(gv);
  float cn = fv * c[idx] + iv * gv;
  c[idx] = cn;
  unsigned short hb = f32_to_bf16_bits(ov * tanhf(cn));
  h1[(size_t)b * ld1 + j] = hb;
  h2[(size_t)b * ld2 + j] = hb;
}

// ---------------------------------------------------------------------------
// Blend + stage: compute new action, write output[b,t,:], stage bf16 action
// and x_{t+1} into A0 for the next step. 16384 threads.
// ---------------------------------------------------------------------------
__global__ void blend_k(const float* __restrict__ fcout,
                        const float* __restrict__ x,
                        float* __restrict__ action,
                        unsigned short* __restrict__ A0,
                        float* __restrict__ out, int t) {
  int idx = blockIdx.x * blockDim.x + threadIdx.x;   // [0, 64*256)
  int b = idx >> 8, f = idx & 255;
  float xt = x[((size_t)b * Sq + t) * Fq + f];
  float ov = fcout[idx];
  float av = action[idx];
  // denom = 1 + 0.5*(0.3+0.3+0.4) = 1.5 ; coeffs 1.15*x + 0.15*a + 0.20*out
  float nv = (t == 0) ? xt
                      : (1.15f * xt + 0.15f * av + 0.20f * ov) * (1.0f / 1.5f);
  out[((size_t)b * Sq + t) * Fq + f] = nv;
  action[idx] = nv;
  A0[(size_t)b * K0q + Fq + f] = f32_to_bf16_bits(nv);
  if (t + 1 < Sq)
    A0[(size_t)b * K0q + f] =
        f32_to_bf16_bits(x[((size_t)b * Sq + t + 1) * Fq + f]);
}

// ------------------------- one-time prep kernels ---------------------------
__global__ void concat_w_k(const float* __restrict__ Wa, int Ka,
                           const float* __restrict__ Wb, int Kb,
                           unsigned short* __restrict__ dst, int total) {
  int idx = blockIdx.x * blockDim.x + threadIdx.x;
  if (idx >= total) return;
  int K = Ka + Kb;
  int row = idx / K, k = idx - row * K;
  float v = (k < Ka) ? Wa[(size_t)row * Ka + k]
                     : Wb[(size_t)row * Kb + (k - Ka)];
  dst[idx] = f32_to_bf16_bits(v);
}

__global__ void conv_w_k(const float* __restrict__ src,
                         unsigned short* __restrict__ dst, int n) {
  int idx = blockIdx.x * blockDim.x + threadIdx.x;
  if (idx < n) dst[idx] = f32_to_bf16_bits(src[idx]);
}

__global__ void addb_k(const float* __restrict__ a, const float* __restrict__ b,
                       float* __restrict__ dst, int n) {
  int idx = blockIdx.x * blockDim.x + threadIdx.x;
  if (idx < n) dst[idx] = a[idx] + b[idx];
}

__global__ void zero_u16_k(unsigned short* p, int n) {
  int idx = blockIdx.x * blockDim.x + threadIdx.x;
  if (idx < n) p[idx] = 0;
}

__global__ void zero_f32_k(float* p, int n) {
  int idx = blockIdx.x * blockDim.x + threadIdx.x;
  if (idx < n) p[idx] = 0.f;
}

__global__ void stage_x0_k(const float* __restrict__ x,
                           unsigned short* __restrict__ A0) {
  int idx = blockIdx.x * blockDim.x + threadIdx.x;   // [0, 64*256)
  int b = idx >> 8, f = idx & 255;
  A0[(size_t)b * K0q + f] = f32_to_bf16_bits(x[(size_t)b * Sq * Fq + f]);
}

// ---------------------------------------------------------------------------
extern "C" void kernel_launch(void* const* d_in, const int* in_sizes, int n_in,
                              void* d_out, int out_size, void* d_ws,
                              size_t ws_size, hipStream_t stream) {
  (void)in_sizes; (void)n_in; (void)out_size; (void)ws_size;
  const float* x    = (const float*)d_in[0];
  const float* Wih0 = (const float*)d_in[1];
  const float* Whh0 = (const float*)d_in[2];
  const float* bih0 = (const float*)d_in[3];
  const float* bhh0 = (const float*)d_in[4];
  const float* Wih1 = (const float*)d_in[5];
  const float* Whh1 = (const float*)d_in[6];
  const float* bih1 = (const float*)d_in[7];
  const float* bhh1 = (const float*)d_in[8];
  const float* fcW  = (const float*)d_in[9];
  const float* fcb  = (const float*)d_in[10];
  float* out = (float*)d_out;

  char* ws = (char*)d_ws;
  size_t off = 0;
  auto carve = [&](size_t bytes) -> char* {
    char* p = ws + off;
    off += (bytes + 255) & ~(size_t)255;
    return p;
  };
  unsigned short* Wc0  = (unsigned short*)carve((size_t)G4q * K0q * 2); // 12.6MB
  unsigned short* Wc1  = (unsigned short*)carve((size_t)G4q * K1q * 2); // 16.8MB
  unsigned short* fcWb = (unsigned short*)carve((size_t)Fq * Hq * 2);
  float* bsum0  = (float*)carve((size_t)G4q * 4);
  float* bsum1  = (float*)carve((size_t)G4q * 4);
  unsigned short* A0 = (unsigned short*)carve((size_t)Bq * K0q * 2);
  unsigned short* A1 = (unsigned short*)carve((size_t)Bq * K1q * 2);
  float* gates0 = (float*)carve((size_t)Bq * G4q * 4);
  float* gates1 = (float*)carve((size_t)Bq * G4q * 4);
  float* fcout  = (float*)carve((size_t)Bq * Fq * 4);
  float* c0     = (float*)carve((size_t)Bq * Hq * 4);
  float* c1     = (float*)carve((size_t)Bq * Hq * 4);
  float* act    = (float*)carve((size_t)Bq * Fq * 4);

  dim3 blk(256);
  // ---- one-time prep (captured into the graph; deterministic) ----
  { int n = G4q * K0q;
    concat_w_k<<<(n + 255) / 256, blk, 0, stream>>>(Wih0, 2 * Fq, Whh0, Hq, Wc0, n); }
  { int n = G4q * K1q;
    concat_w_k<<<(n + 255) / 256, blk, 0, stream>>>(Wih1, Hq, Whh1, Hq, Wc1, n); }
  { int n = Fq * Hq;
    conv_w_k<<<(n + 255) / 256, blk, 0, stream>>>(fcW, fcWb, n); }
  addb_k<<<(G4q + 255) / 256, blk, 0, stream>>>(bih0, bhh0, bsum0, G4q);
  addb_k<<<(G4q + 255) / 256, blk, 0, stream>>>(bih1, bhh1, bsum1, G4q);
  { int n = Bq * K0q; zero_u16_k<<<(n + 255) / 256, blk, 0, stream>>>(A0, n); }
  { int n = Bq * K1q; zero_u16_k<<<(n + 255) / 256, blk, 0, stream>>>(A1, n); }
  zero_f32_k<<<(Bq * Hq + 255) / 256, blk, 0, stream>>>(c0, Bq * Hq);
  zero_f32_k<<<(Bq * Hq + 255) / 256, blk, 0, stream>>>(c1, Bq * Hq);
  zero_f32_k<<<(Bq * Fq + 255) / 256, blk, 0, stream>>>(act, Bq * Fq);
  stage_x0_k<<<(Bq * Fq) / 256, blk, 0, stream>>>(x, A0);

  // ---- sequential recurrence: 256 steps, stream order carries the deps ----
  for (int t = 0; t < Sq; ++t) {
    // layer 0: gates0 = [x_t|action|h0] @ Wc0^T + bsum0
    gemm64_wmma_bf16<<<G4q / 64, blk, 0, stream>>>(
        (const __bf16*)A0, K0q, (const __bf16*)Wc0, K0q, bsum0, gates0, G4q);
    // h0 -> A1[:,0:1024] and A0[:,512:1536]
    lstm_cell_k<<<(Bq * Hq) / 256, blk, 0, stream>>>(
        gates0, c0, A1, K1q, A0 + 2 * Fq, K0q);
    // layer 1: gates1 = [h0|h1] @ Wc1^T + bsum1
    gemm64_wmma_bf16<<<G4q / 64, blk, 0, stream>>>(
        (const __bf16*)A1, K1q, (const __bf16*)Wc1, K1q, bsum1, gates1, G4q);
    // h1 -> A1[:,1024:2048] (written once; second dest aliases harmlessly)
    lstm_cell_k<<<(Bq * Hq) / 256, blk, 0, stream>>>(
        gates1, c1, A1 + Hq, K1q, A1 + Hq, K1q);
    // fc: out = h1 @ fcW^T + fcb  (A = A1 col-offset 1024, lda = 2048)
    gemm64_wmma_bf16<<<Fq / 64, blk, 0, stream>>>(
        (const __bf16*)(A1 + Hq), K1q, (const __bf16*)fcWb, Hq, fcb, fcout, Fq);
    // blend, emit action_t, stage action & x_{t+1} for next step
    blend_k<<<(Bq * Fq) / 256, blk, 0, stream>>>(fcout, x, act, A0, out, t);
  }
}
